// GraphRanker_36464272343476
// MI455X (gfx1250) — compile-verified, matrix operand order checked by reading, not run on hardware
//
#include <hip/hip_runtime.h>
#include <hip/hip_bf16.h>

typedef __attribute__((ext_vector_type(16))) _Float16 v16h;
typedef __attribute__((ext_vector_type(8)))  _Float16 v8h;
typedef __attribute__((ext_vector_type(8)))  float    v8f;
typedef __attribute__((ext_vector_type(4)))  unsigned int v4u;
typedef __attribute__((ext_vector_type(8)))  int      v8i;
typedef __attribute__((ext_vector_type(4)))  int      v4i;

#define NNODE 40000
#define NEDGE 320000
#define NBATCH 64
#define DDIM 256
#define HIDD 128

// ---------------------------------------------------------------------------
// Pack an f32 row-major [K,M] weight into the CDNA5 16x32-f16 B-fragment
// layout: [kt][nt][lane 0..31][half 0..15], so each lane reads 2 x b128.
// Lane mapping (ISA 7.12.2, 16-bit 16x32): lanes 0-15 halves 0..7 = K0..7,
// halves 8..15 = K16..23; lanes 16-31 halves 0..7 = K8..15, 8..15 = K24..31.
// ---------------------------------------------------------------------------
__global__ void pack_b_kernel(const float* __restrict__ W, _Float16* __restrict__ Bp,
                              int K, int M) {
    int idx = blockIdx.x * blockDim.x + threadIdx.x;
    if (idx >= K * M) return;
    int k = idx / M, m = idx - k * M;
    int kt = k >> 5, kin = k & 31;
    int nt = m >> 4, col = m & 15;
    int grp = kin >> 3;
    int j8  = kin & 7;
    int lhi = grp & 1;
    int j   = ((grp >> 1) << 3) + j8;
    int lane = (lhi << 4) + col;
    size_t dst = (((size_t)(kt * (M >> 4) + nt)) * 32 + lane) * 16 + j;
    Bp[dst] = (_Float16)W[idx];
}

// ---------------------------------------------------------------------------
// WMMA GEMM with TDM-staged B panels.
// Block = 128 thr = 4 waves; block tile 64 rows x 64 cols; wave tile 16x64.
// Per k-step the 4 KB B panel (4 nt x 32 lanes x 16 halves) is DMA'd into LDS
// by the Tensor Data Mover (double-buffered, s_wait_tensorcnt at distance 1);
// waves read fragments from LDS via ds_load_b128. A loads stream from global.
// K mult of 32, M mult of 64, rows mult of 64. Dynamic LDS = 8192 bytes.
// ---------------------------------------------------------------------------
template <int ACT> // 0 = none, 1 = relu
__global__ void gemm_wmma(const _Float16* __restrict__ A,
                          const _Float16* __restrict__ Bp,
                          const float* __restrict__ bias,
                          float* __restrict__ Cf,
                          _Float16* __restrict__ Ch,
                          int K, int M) {
    extern __shared__ _Float16 smemB[];     // 2 x 4KB panels
    const int wave = threadIdx.x >> 5;
    const int lane = threadIdx.x & 31;
    const int hi   = lane >> 4;
    const int mrow = lane & 15;
    const int row0 = blockIdx.x * 64 + wave * 16;
    const int col0 = blockIdx.y * 64;
    const int MT   = M >> 4;
    const int nt0  = col0 >> 4;

    // TDM D# group1 (constant): 1-D copy, data_size=8B (code 3),
    // tile_dim0 = tensor_dim0 = stride = 512 elements (4096 bytes).
    v8i g1;
    g1[0] = 0x00030000;   // workgroup_mask=0, data_size=3, no flags
    g1[1] = 0x02000000;   // tensor_dim0[15:0]=512 in bits 63:48
    g1[2] = 0x00010000;   // tensor_dim1=1
    g1[3] = 0x02000000;   // tile_dim0=512 in bits 127:112
    g1[4] = 0;            // tile_dim1=0, tile_dim2=0
    g1[5] = 512;          // tensor_dim0_stride[31:0]
    g1[6] = 0;
    g1[7] = 0;
    v4i z4;
    z4[0] = z4[1] = z4[2] = z4[3] = 0;
    v8i z8;
    z8[0] = z8[1] = z8[2] = z8[3] = z8[4] = z8[5] = z8[6] = z8[7] = 0;

    const int ksteps = K >> 5;

    // prologue: DMA panel 0 into LDS buffer 0
    if (threadIdx.x < 32) {
        unsigned long long ga =
            (unsigned long long)(const void*)(Bp + ((size_t)(0 * MT + nt0) * 32) * 16);
        v4u g0;
        g0[0] = 1u;                                   // count=1
        g0[1] = 0u;                                   // lds_addr = 0
        g0[2] = (unsigned)ga;                         // global_addr lo
        g0[3] = (unsigned)(ga >> 32) | (2u << 30);    // global_addr hi | type=2
        __builtin_amdgcn_tensor_load_to_lds(g0, g1, z4, z4, z8, 0);
    }

    v8f acc[4] = {};
    for (int kt = 0; kt < ksteps; ++kt) {
        if (threadIdx.x < 32) {
            if (kt + 1 < ksteps) {
                unsigned long long ga = (unsigned long long)(const void*)
                    (Bp + ((size_t)((kt + 1) * MT + nt0) * 32) * 16);
                v4u g0;
                g0[0] = 1u;
                g0[1] = (unsigned)(((kt + 1) & 1) * 4096);   // alternate panel
                g0[2] = (unsigned)ga;
                g0[3] = (unsigned)(ga >> 32) | (2u << 30);
                __builtin_amdgcn_tensor_load_to_lds(g0, g1, z4, z4, z8, 0);
                __builtin_amdgcn_s_wait_tensorcnt(1);  // panel kt complete
            } else {
                __builtin_amdgcn_s_wait_tensorcnt(0);
            }
        }
        __syncthreads();                               // panel kt visible to all

        // A fragment: row (row0+mrow), 32 K-values starting kt*32
        const _Float16* arow = A + (size_t)(row0 + mrow) * K + (kt << 5);
        v8h alo = *(const v8h*)(arow + hi * 8);
        v8h ahi = *(const v8h*)(arow + 16 + hi * 8);
        v16h a;
#pragma unroll
        for (int j = 0; j < 8; ++j) { a[j] = alo[j]; a[8 + j] = ahi[j]; }

        const _Float16* panel = smemB + (kt & 1) * 2048;  // halves
#pragma unroll
        for (int t = 0; t < 4; ++t) {
            const _Float16* bptr = panel + ((t * 32 + lane) << 4);
            v8h blo = *(const v8h*)(bptr);
            v8h bhi = *(const v8h*)(bptr + 8);
            v16h b;
#pragma unroll
            for (int j = 0; j < 8; ++j) { b[j] = blo[j]; b[8 + j] = bhi[j]; }
            acc[t] = __builtin_amdgcn_wmma_f32_16x16x32_f16(
                false, a, false, b, (short)0, acc[t], false, false);
        }
        __syncthreads();   // done reading panel before it is overwritten (kt+2)
    }

    // C layout: VGPR r -> row = row0 + hi*8 + r ; col = col0 + t*16 + mrow
#pragma unroll
    for (int t = 0; t < 4; ++t) {
        int col = col0 + t * 16 + mrow;
        float bv = bias ? bias[col] : 0.0f;
#pragma unroll
        for (int r = 0; r < 8; ++r) {
            int row = row0 + hi * 8 + r;
            float v = acc[t][r] + bv;
            if (ACT == 1) v = fmaxf(v, 0.0f);
            size_t o = (size_t)row * M + col;
            if (Cf) Cf[o] = v;
            if (Ch) Ch[o] = (_Float16)v;
        }
    }
}

// ---------------------------------------------------------------------------
// Elementwise / graph kernels
// ---------------------------------------------------------------------------
__global__ void prep_kernel(const float* __restrict__ x, const float* __restrict__ query,
                            const int* __restrict__ batch,
                            _Float16* __restrict__ g_in, _Float16* __restrict__ q16,
                            _Float16* __restrict__ inter16) {
    int idx = blockIdx.x * blockDim.x + threadIdx.x;
    if (idx >= NNODE * DDIM) return;
    int n = idx >> 8, d = idx & 255;
    int b = batch[n];
    float q  = query[b * DDIM + d];
    float xv = x[idx];
    g_in[(size_t)n * 512 + d]       = (_Float16)xv;
    g_in[(size_t)n * 512 + 256 + d] = (_Float16)q;
    q16[idx]     = (_Float16)q;
    inter16[idx] = (_Float16)(xv * q);
}

__global__ void gate_epilogue(const float* __restrict__ logits, const float* __restrict__ x,
                              const _Float16* __restrict__ q16,
                              const _Float16* __restrict__ inter16,
                              float* __restrict__ x0, _Float16* __restrict__ hA) {
    int idx = blockIdx.x * blockDim.x + threadIdx.x;
    if (idx >= NNODE * DDIM) return;
    int n = idx >> 8, d = idx & 255;
    float g = 1.0f / (1.0f + __expf(-logits[idx]));
    float v = x[idx] * g;
    x0[idx] = v;
    size_t base = (size_t)n * 768;
    hA[base + d]       = (_Float16)v;
    hA[base + 256 + d] = q16[idx];
    hA[base + 512 + d] = inter16[idx];
}

// per (relation,node): alphaD = xw @ Q, alphaS = xw @ K
__global__ void node_alpha(const _Float16* __restrict__ xw,
                           const float* __restrict__ Qw, const float* __restrict__ Kw,
                           float* __restrict__ aD, float* __restrict__ aS,
                           int M, int H) {
    int idx = blockIdx.x * blockDim.x + threadIdx.x;
    if (idx >= 2 * NNODE) return;
    const _Float16* xr = xw + (size_t)idx * M;
    float accq[4] = {0.f, 0.f, 0.f, 0.f};
    float acck[4] = {0.f, 0.f, 0.f, 0.f};
    for (int m = 0; m < M; ++m) {
        float xv = (float)xr[m];
        for (int h = 0; h < H; ++h) {
            accq[h] = fmaf(xv, Qw[m * H + h], accq[h]);
            acck[h] = fmaf(xv, Kw[m * H + h], acck[h]);
        }
    }
    for (int h = 0; h < H; ++h) {
        aD[(size_t)idx * H + h] = accq[h];
        aS[(size_t)idx * H + h] = acck[h];
    }
}

__global__ void fill_f32(float* __restrict__ p, float v, size_t count) {
    size_t stride = (size_t)gridDim.x * blockDim.x;
    for (size_t i = blockIdx.x * (size_t)blockDim.x + threadIdx.x; i < count; i += stride)
        p[i] = v;
}

__device__ inline void atomicMaxFloat(float* addr, float v) {
    if (v >= 0.0f) atomicMax((int*)addr, __float_as_int(v));
    else           atomicMin((unsigned int*)addr, (unsigned int)__float_as_int(v));
}

__global__ void edge_alpha_max(const int* __restrict__ ei, const int* __restrict__ et,
                               const float* __restrict__ aD, const float* __restrict__ aS,
                               float* __restrict__ atmp, float* __restrict__ amax, int H) {
    int e = blockIdx.x * blockDim.x + threadIdx.x;
    if (e >= NEDGE) return;
    int s = ei[e], d = ei[NEDGE + e], t = et[e];
    for (int h = 0; h < H; ++h) {
        float a = aD[((size_t)t * NNODE + d) * H + h] + aS[((size_t)t * NNODE + s) * H + h];
        a = (a > 0.0f) ? a : 0.2f * a;     // leaky_relu(0.2)
        atmp[(size_t)e * H + h] = a;
        atomicMaxFloat(&amax[(size_t)d * H + h], a);
    }
}

__global__ void edge_expsum(const int* __restrict__ ei,
                            float* __restrict__ atmp, const float* __restrict__ amax,
                            float* __restrict__ denom, int H) {
    int e = blockIdx.x * blockDim.x + threadIdx.x;
    if (e >= NEDGE) return;
    int d = ei[NEDGE + e];
    for (int h = 0; h < H; ++h) {
        float ex = __expf(atmp[(size_t)e * H + h] - amax[(size_t)d * H + h]);
        atmp[(size_t)e * H + h] = ex;
        atomicAdd(&denom[(size_t)d * H + h], ex);
    }
}

// one wave per edge; out_ch is always 128 so head = col >> 7
__global__ void edge_message(const int* __restrict__ ei, const int* __restrict__ et,
                             const _Float16* __restrict__ xw,
                             const float* __restrict__ atmp, const float* __restrict__ denom,
                             float* __restrict__ agg, int M, int H) {
    int gw   = (blockIdx.x * blockDim.x + threadIdx.x) >> 5;
    int lane = threadIdx.x & 31;
    if (gw >= NEDGE) return;
    int s = ei[gw], d = ei[NEDGE + gw], t = et[gw];
    const _Float16* row = xw + ((size_t)t * NNODE + s) * M;
    for (int c = lane; c < M; c += 32) {
        int h = c >> 7;
        float w = atmp[(size_t)gw * H + h] / (denom[(size_t)d * H + h] + 1e-16f);
        atomicAdd(&agg[(size_t)d * M + c], (float)row[c] * w);
    }
}

__global__ void node_out(const float* __restrict__ agg, const float* __restrict__ bias,
                         const _Float16* __restrict__ resid, _Float16* __restrict__ hout,
                         int total, int M) {
    int idx = blockIdx.x * blockDim.x + threadIdx.x;
    if (idx >= total) return;
    int c = idx % M;
    float v = fmaxf(agg[idx] + bias[c], 0.0f);
    if (resid) v += (float)resid[idx];
    hout[idx] = (_Float16)v;
}

// one wave per node: out = (out + x0) / max(||.||, 1e-12), in place on d_out
__global__ void finalize_kernel(float* __restrict__ out, const float* __restrict__ x0) {
    int gw   = (blockIdx.x * blockDim.x + threadIdx.x) >> 5;
    int lane = threadIdx.x & 31;
    if (gw >= NNODE) return;
    size_t base = (size_t)gw * DDIM;
    float vals[8];
    float ss = 0.0f;
#pragma unroll
    for (int i = 0; i < 8; ++i) {
        int c = lane + i * 32;
        float v = out[base + c] + x0[base + c];
        vals[i] = v;
        ss = fmaf(v, v, ss);
    }
#pragma unroll
    for (int off = 16; off > 0; off >>= 1) ss += __shfl_xor(ss, off, 32);
    float inv = 1.0f / fmaxf(sqrtf(ss), 1e-12f);
#pragma unroll
    for (int i = 0; i < 8; ++i) out[base + lane + i * 32] = vals[i] * inv;
}

// ---------------------------------------------------------------------------
extern "C" void kernel_launch(void* const* d_in, const int* in_sizes, int n_in,
                              void* d_out, int out_size, void* d_ws, size_t ws_size,
                              hipStream_t stream) {
    const float* x       = (const float*)d_in[0];
    const float* query   = (const float*)d_in[1];
    const int*   ei      = (const int*)d_in[2];
    const int*   etype   = (const int*)d_in[3];
    const int*   batch   = (const int*)d_in[4];
    const float* gate_w1 = (const float*)d_in[5];
    const float* gate_b1 = (const float*)d_in[6];
    const float* gate_w2 = (const float*)d_in[7];
    const float* gate_b2 = (const float*)d_in[8];
    const float* W1      = (const float*)d_in[9];
    const float* q1      = (const float*)d_in[10];
    const float* k1      = (const float*)d_in[11];
    const float* b1      = (const float*)d_in[12];
    const float* W2      = (const float*)d_in[13];
    const float* q2      = (const float*)d_in[14];
    const float* k2      = (const float*)d_in[15];
    const float* b2      = (const float*)d_in[16];
    const float* W3      = (const float*)d_in[17];
    const float* q3      = (const float*)d_in[18];
    const float* k3      = (const float*)d_in[19];
    const float* b3      = (const float*)d_in[20];
    const float* proj_w  = (const float*)d_in[21];
    const float* proj_b  = (const float*)d_in[22];

    // ---- carve workspace (256B aligned chunks) ----
    char* base = (char*)d_ws;
    size_t off = 0;
    auto carve = [&](size_t bytes) -> void* {
        off = (off + 255) & ~(size_t)255;
        void* p = base + off;
        off += bytes;
        return p;
    };
    const size_t N = NNODE, E = NEDGE;
    _Float16* pw_gw1  = (_Float16*)carve(512 * 256 * 2);
    _Float16* pw_gw2  = (_Float16*)carve(256 * 256 * 2);
    _Float16* pw_w1a  = (_Float16*)carve(768 * 512 * 2);
    _Float16* pw_w1b  = (_Float16*)carve(768 * 512 * 2);
    _Float16* pw_w2a  = (_Float16*)carve(512 * 512 * 2);
    _Float16* pw_w2b  = (_Float16*)carve(512 * 512 * 2);
    _Float16* pw_w3a  = (_Float16*)carve(512 * 128 * 2);
    _Float16* pw_w3b  = (_Float16*)carve(512 * 128 * 2);
    _Float16* pw_proj = (_Float16*)carve(128 * 256 * 2);
    _Float16* q16     = (_Float16*)carve(N * 256 * 2);
    _Float16* inter16 = (_Float16*)carve(N * 256 * 2);
    _Float16* g_in16  = (_Float16*)carve(N * 512 * 2);   // aliased as logits f32 later
    _Float16* gh16    = (_Float16*)carve(N * 256 * 2);
    float*    x0      = (float*)carve(N * 256 * 4);
    _Float16* hA      = (_Float16*)carve(N * 768 * 2);
    _Float16* hB      = (_Float16*)carve(N * 512 * 2);
    _Float16* xw      = (_Float16*)carve(2 * N * 512 * 2);
    float*    aD      = (float*)carve(2 * N * 4 * 4);
    float*    aS      = (float*)carve(2 * N * 4 * 4);
    float*    amax    = (float*)carve(N * 4 * 4);
    float*    denom   = (float*)carve(N * 4 * 4);
    float*    atmp    = (float*)carve(E * 4 * 4);
    float*    agg     = (float*)carve(N * 512 * 4);
    float*    logits  = (float*)g_in16;   // reuse: g_in dead after stage 1

    auto cdiv = [](long long a, long long b) { return (unsigned)((a + b - 1) / b); };
    const size_t SMEM = 8192;   // 2 x 4KB TDM B panels

    // ---- pack all GEMM weights to f16 B-fragment layout ----
    pack_b_kernel<<<cdiv(512 * 256, 256), 256, 0, stream>>>(gate_w1, pw_gw1, 512, 256);
    pack_b_kernel<<<cdiv(256 * 256, 256), 256, 0, stream>>>(gate_w2, pw_gw2, 256, 256);
    pack_b_kernel<<<cdiv(768 * 512, 256), 256, 0, stream>>>(W1,             pw_w1a, 768, 512);
    pack_b_kernel<<<cdiv(768 * 512, 256), 256, 0, stream>>>(W1 + 768 * 512, pw_w1b, 768, 512);
    pack_b_kernel<<<cdiv(512 * 512, 256), 256, 0, stream>>>(W2,             pw_w2a, 512, 512);
    pack_b_kernel<<<cdiv(512 * 512, 256), 256, 0, stream>>>(W2 + 512 * 512, pw_w2b, 512, 512);
    pack_b_kernel<<<cdiv(512 * 128, 256), 256, 0, stream>>>(W3,             pw_w3a, 512, 128);
    pack_b_kernel<<<cdiv(512 * 128, 256), 256, 0, stream>>>(W3 + 512 * 128, pw_w3b, 512, 128);
    pack_b_kernel<<<cdiv(128 * 256, 256), 256, 0, stream>>>(proj_w, pw_proj, 128, 256);

    // ---- stage 0: gather q_node, build gate input + interaction ----
    prep_kernel<<<cdiv(N * 256, 256), 256, 0, stream>>>(x, query, batch, g_in16, q16, inter16);

    // ---- gate MLP ----
    dim3 blk(128);
    gemm_wmma<1><<<dim3(N / 64, 256 / 64), blk, SMEM, stream>>>(g_in16, pw_gw1, gate_b1,
                                                                nullptr, gh16, 512, 256);
    gemm_wmma<0><<<dim3(N / 64, 256 / 64), blk, SMEM, stream>>>(gh16, pw_gw2, gate_b2,
                                                                logits, nullptr, 256, 256);
    gate_epilogue<<<cdiv(N * 256, 256), 256, 0, stream>>>(logits, x, q16, inter16, x0, hA);

    // ---- generic RGAT layer launcher ----
    auto rgat = [&](const _Float16* hin, int K, int M, int H,
                    const _Float16* pwa, const _Float16* pwb,
                    const float* Qw, const float* Kw, const float* bias,
                    const _Float16* resid, _Float16* hout) {
        gemm_wmma<0><<<dim3(N / 64, M / 64), blk, SMEM, stream>>>(hin, pwa, nullptr,
                                                                  nullptr, xw, K, M);
        gemm_wmma<0><<<dim3(N / 64, M / 64), blk, SMEM, stream>>>(hin, pwb, nullptr,
                                                                  nullptr, xw + N * M, K, M);
        node_alpha<<<cdiv(2 * N, 256), 256, 0, stream>>>(xw, Qw, Kw, aD, aS, M, H);
        fill_f32<<<1024, 256, 0, stream>>>(amax, -1e30f, N * H);
        fill_f32<<<1024, 256, 0, stream>>>(denom, 0.0f, N * H);
        fill_f32<<<2048, 256, 0, stream>>>(agg, 0.0f, N * (size_t)M);
        edge_alpha_max<<<cdiv(E, 256), 256, 0, stream>>>(ei, etype, aD, aS, atmp, amax, H);
        edge_expsum<<<cdiv(E, 256), 256, 0, stream>>>(ei, atmp, amax, denom, H);
        edge_message<<<cdiv(E * 32, 256), 256, 0, stream>>>(ei, etype, xw, atmp, denom,
                                                            agg, M, H);
        node_out<<<cdiv(N * (size_t)M, 256), 256, 0, stream>>>(agg, bias, resid, hout,
                                                               (int)(N * M), M);
    };

    rgat(hA, 768, 512, 4, pw_w1a, pw_w1b, q1, k1, b1, nullptr, hB);   // h1 -> hB
    rgat(hB, 512, 512, 4, pw_w2a, pw_w2b, q2, k2, b2, hB, hA);        // h2 = h1 + relu(.)
    rgat(hA, 512, 128, 1, pw_w3a, pw_w3b, q3, k3, b3, nullptr, hB);   // h3 -> hB

    // ---- projection + residual + L2 normalize ----
    float* out = (float*)d_out;
    gemm_wmma<0><<<dim3(N / 64, 256 / 64), blk, SMEM, stream>>>(hB, pw_proj, proj_b,
                                                                out, nullptr, 128, 256);
    finalize_kernel<<<cdiv(N * 32, 256), 256, 0, stream>>>(out, x0);
}
